// HybridQNetwork_78847009620359
// MI455X (gfx1250) — matrix operand order repaired and checked by reading, not compiled, not run on hardware
//
#include <hip/hip_runtime.h>
#include <hip/hip_bf16.h>
#include <hip/hip_fp16.h>

typedef __attribute__((ext_vector_type(16))) _Float16 v16h;
typedef __attribute__((ext_vector_type(8)))  _Float16 v8h;
typedef __attribute__((ext_vector_type(8)))  float    v8f;

__device__ __forceinline__ v16h zero16() {
  v16h z;
#pragma unroll
  for (int i = 0; i < 16; ++i) z[i] = (_Float16)0.f;
  return z;
}

// ---------------------------------------------------------------------------
// BN fold: scale = g * rsqrt(v + eps), shift = b - m*scale
// ---------------------------------------------------------------------------
__global__ void k_bnfold(const float* __restrict__ g, const float* __restrict__ b,
                         const float* __restrict__ m, const float* __restrict__ v,
                         float* __restrict__ scale, float* __restrict__ shift, int C) {
  int i = blockIdx.x * blockDim.x + threadIdx.x;
  if (i < C) {
    float inv = g[i] * rsqrtf(v[i] + 1e-5f);
    scale[i] = inv;
    shift[i] = b[i] - m[i] * inv;
  }
}

// ---------------------------------------------------------------------------
// Weight convert: OIHW f32 -> [cout][(kh*KW+kw)*Cin + cin] f16 (row = GEMM K)
// ---------------------------------------------------------------------------
__global__ void k_wconv(const float* __restrict__ w, _Float16* __restrict__ out,
                        int O, int I, int KH, int KW) {
  int idx = blockIdx.x * blockDim.x + threadIdx.x;
  int Kk = I * KH * KW;
  int total = O * Kk;
  if (idx >= total) return;
  int o = idx / Kk;
  int r = idx % Kk;
  int ci = r % I;
  int q = r / I;
  int kw = q % KW;
  int kh = q / KW;
  out[idx] = (_Float16)w[((size_t)(o * I + ci) * KH + kh) * KW + kw];
}

// ---------------------------------------------------------------------------
// NCHW f32 -> NHWC f16
// ---------------------------------------------------------------------------
__global__ void k_nchw2nhwc(const float* __restrict__ in, _Float16* __restrict__ out,
                            int N, int C, int H, int W) {
  int idx = blockIdx.x * blockDim.x + threadIdx.x;
  int total = N * C * H * W;
  if (idx >= total) return;
  int c = idx % C;
  int t = idx / C;
  int w = t % W; t /= W;
  int h = t % H;
  int n = t / H;
  out[idx] = (_Float16)in[((size_t)(n * C + c) * H + h) * W + w];
}

// ---------------------------------------------------------------------------
// Main implicit-GEMM conv, Cin % 32 == 0, Cout % 64 == 0.
// Block: 128 threads (4 waves). Tile: 64 pixels x 64 couts.
// Wave computes 4 x (16 pixels x 16 couts) accumulators; one shared B fragment
// per k-step feeds 4 WMMAs. A-panel staged through LDS (double buffered) and
// shared by all 4 waves. K consumed as (kh,kw) taps x 32-cin blocks: no
// integer division anywhere in the hot loop, and only 4 divisions (one per
// subtile) in the epilogue via carry-incremented (ow,oh,nn).
// ---------------------------------------------------------------------------
__global__ void __launch_bounds__(128)
k_conv_wmma(const _Float16* __restrict__ in,     // NHWC f16
            const _Float16* __restrict__ wt,     // [Cout][K]
            const float* __restrict__ scale,
            const float* __restrict__ shift,
            const _Float16* __restrict__ residual,
            _Float16* __restrict__ out,          // NHWC f16
            int N, int Hin, int Win, int Cin,
            int Cout, int Hout, int Wout,
            int KH, int KW, int stride, int pad, int relu) {
  __shared__ _Float16 smem[2][64 * 32];          // double-buffered A panel (2x4KB)

  const int tid  = threadIdx.x;
  const int lane = tid & 31;
  const int wave = tid >> 5;
  const int mBase = blockIdx.x * 64;
  const int coutBase = (blockIdx.y * 4 + wave) * 16;  // Cout % 64 == 0: always valid

  const int M  = N * Hout * Wout;
  const int K  = KH * KW * Cin;
  const int HW = Hout * Wout;

  const int row  = lane & 15;                    // A row within subtile / B,D column
  const int half = lane >> 4;
  const int nCol = coutBase + row;

  // -- cooperative A-staging role: thread stages pixel cp, 16-f16 half chalf
  const int cp = tid >> 1;
  const int chalf = tid & 1;
  int cpix = mBase + cp;
  bool cvalidPix = cpix < M;
  int cn = 0, cho = 0, cwo = 0;
  if (cvalidPix) {
    cn = cpix / HW;
    int r2 = cpix % HW;
    cho = r2 / Wout;
    cwo = r2 % Wout;
  }
  const int chb = cho * stride - pad;
  const int cwb = cwo * stride - pad;

  const _Float16* wrow = wt + (size_t)nCol * K;

  v8f acc[4];
#pragma unroll
  for (int s = 0; s < 4; ++s)
#pragma unroll
    for (int r = 0; r < 8; ++r) acc[s][r] = 0.f;

  int ph = 0;
  for (int kh = 0; kh < KH; ++kh) {
    for (int kw = 0; kw < KW; ++kw) {
      const int ch = chb + kh;
      const int cw = cwb + kw;
      const bool cv = cvalidPix && (unsigned)ch < (unsigned)Hin && (unsigned)cw < (unsigned)Win;
      const _Float16* cptr = in + (((size_t)cn * Hin + ch) * Win + cw) * Cin + chalf * 16;
      const int kbase = (kh * KW + kw) * Cin;

      for (int c0 = 0; c0 < Cin; c0 += 32) {
        // stage 64x32 A block into LDS (each thread: 32 contiguous bytes)
        v16h av = cv ? *(const v16h*)(cptr + c0) : zero16();
        *(v16h*)&smem[ph][cp * 32 + chalf * 16] = av;
        __syncthreads();

        // B fragment: 16 contiguous K values of this lane's cout column
        v16h b = *(const v16h*)(wrow + kbase + c0 + (half << 4));
        __builtin_prefetch((const void*)(wrow + kbase + c0 + (half << 4) + 32), 0, 1);

#pragma unroll
        for (int s = 0; s < 4; ++s) {
          const _Float16* lp = &smem[ph][(s * 16 + row) * 32 + (half << 4)];
          v8h r0 = *(const v8h*)lp;
          v8h r1 = *(const v8h*)(lp + 8);
          v16h a = __builtin_shufflevector(r0, r1, 0, 1, 2, 3, 4, 5, 6, 7,
                                           8, 9, 10, 11, 12, 13, 14, 15);
          acc[s] = __builtin_amdgcn_wmma_f32_16x16x32_f16(
              false, a, false, b, (short)0, acc[s], false, false);
        }
        ph ^= 1;
      }
    }
  }

  // ---- epilogue: D layout col = lane&15, row m = 8*half + r (per subtile)
  const float scv = scale[nCol];
  const float shv = shift[nCol];
  const bool hasRes = residual != nullptr;
#pragma unroll
  for (int s = 0; s < 4; ++s) {
    const int m0 = mBase + s * 16 + half * 8;
    int nn = m0 / HW;                // one decode per subtile, then carry-add
    int rr = m0 % HW;
    int oh = rr / Wout;
    int ow = rr % Wout;
#pragma unroll
    for (int r = 0; r < 8; ++r) {
      if (m0 + r < M) {
        float v = acc[s][r] * scv + shv;
        size_t oidx = (((size_t)nn * Hout + oh) * Wout + ow) * Cout + nCol;
        if (hasRes) v += (float)residual[oidx];
        if (relu) v = fmaxf(v, 0.f);
        out[oidx] = (_Float16)v;
      }
      if (++ow == Wout) { ow = 0; if (++oh == Hout) { oh = 0; ++nn; } }
    }
  }
}

// ---------------------------------------------------------------------------
// Stem conv: 7x7 s2 p3, Cin=3 -> Cout=64, all divisors compile-time constants.
// One wave = 16 pixels x 16 couts; 4 waves/block cover 64 couts.
// ---------------------------------------------------------------------------
__global__ void __launch_bounds__(128)
k_conv_stem(const _Float16* __restrict__ in,    // NHWC f16, 224x224x3
            const _Float16* __restrict__ wt,    // [64][147]
            const float* __restrict__ scale,
            const float* __restrict__ shift,
            _Float16* __restrict__ out,         // NHWC f16, 112x112x64
            int N) {
  constexpr int Hin = 224, Win = 224, Cin = 3, Cout = 64;
  constexpr int Hout = 112, Wout = 112, KH = 7, KW = 7, stride = 2, pad = 3;
  constexpr int K = KH * KW * Cin;              // 147
  constexpr int HW = Hout * Wout;

  const int lane = threadIdx.x & 31;
  const int wave = threadIdx.x >> 5;
  const int mTile = blockIdx.x;
  const int coutBase = wave * 16;

  const int M = N * HW;
  const int row  = lane & 15;
  const int half = lane >> 4;
  const int nCol = coutBase + row;

  int pix = mTile * 16 + row;
  bool pixValid = pix < M;
  int n = 0, ho = 0, wo = 0;
  if (pixValid) {
    n = pix / HW;
    int r = pix % HW;
    ho = r / Wout;
    wo = r % Wout;
  }
  const int hbase = ho * stride - pad;
  const int wbase = wo * stride - pad;
  const _Float16* wrow = wt + (size_t)nCol * K;

  v8f acc;
#pragma unroll
  for (int r = 0; r < 8; ++r) acc[r] = 0.f;

  for (int k0 = 0; k0 < K; k0 += 32) {
    v16h a;
#pragma unroll
    for (int i = 0; i < 16; ++i) {
      int k = k0 + (half << 4) + i;
      _Float16 val = (_Float16)0.f;
      if (pixValid && k < K) {
        int cin = k % Cin;        // constant divisors -> mul/shift
        int t   = k / Cin;
        int kw  = t % KW;
        int kh  = t / KW;
        int h = hbase + kh;
        int w = wbase + kw;
        if ((unsigned)h < (unsigned)Hin && (unsigned)w < (unsigned)Win)
          val = in[(((size_t)n * Hin + h) * Win + w) * Cin + cin];
      }
      a[i] = val;
    }
    v16h b;
#pragma unroll
    for (int i = 0; i < 16; ++i) {
      int k = k0 + (half << 4) + i;
      b[i] = (k < K) ? wrow[k] : (_Float16)0.f;
    }
    acc = __builtin_amdgcn_wmma_f32_16x16x32_f16(false, a, false, b, (short)0,
                                                 acc, false, false);
  }

  const float scv = scale[nCol];
  const float shv = shift[nCol];
  {
    const int m0 = mTile * 16 + half * 8;
    int nn = m0 / HW;               // constant divisors here too
    int rr = m0 % HW;
    int oh = rr / Wout;
    int ow = rr % Wout;
#pragma unroll
    for (int r = 0; r < 8; ++r) {
      if (m0 + r < M) {
        float v = fmaxf(acc[r] * scv + shv, 0.f);   // stem always ReLUs
        out[(((size_t)nn * Hout + oh) * Wout + ow) * Cout + nCol] = (_Float16)v;
      }
      if (++ow == Wout) { ow = 0; if (++oh == Hout) { oh = 0; ++nn; } }
    }
  }
}

// ---------------------------------------------------------------------------
// 3x3 stride-2 pad-1 maxpool, NHWC f16
// ---------------------------------------------------------------------------
__global__ void k_maxpool(const _Float16* __restrict__ in, _Float16* __restrict__ out,
                          int N, int C, int Hin, int Win, int Hout, int Wout) {
  int idx = blockIdx.x * blockDim.x + threadIdx.x;
  int total = N * Hout * Wout * C;
  if (idx >= total) return;
  int c = idx % C;
  int t = idx / C;
  int wo = t % Wout; t /= Wout;
  int ho = t % Hout;
  int n  = t / Hout;
  float m = -1e30f;
#pragma unroll
  for (int kh = 0; kh < 3; ++kh)
#pragma unroll
    for (int kw = 0; kw < 3; ++kw) {
      int h = ho * 2 - 1 + kh;
      int w = wo * 2 - 1 + kw;
      if ((unsigned)h < (unsigned)Hin && (unsigned)w < (unsigned)Win)
        m = fmaxf(m, (float)in[(((size_t)n * Hin + h) * Win + w) * C + c]);
    }
  out[idx] = (_Float16)m;
}

// ---------------------------------------------------------------------------
// Global average pool NHWC f16 -> [N][C] f32
// ---------------------------------------------------------------------------
__global__ void k_avgpool(const _Float16* __restrict__ in, float* __restrict__ out,
                          int N, int C, int H, int W) {
  int idx = blockIdx.x * blockDim.x + threadIdx.x;
  if (idx >= N * C) return;
  int c = idx % C;
  int b = idx / C;
  float s = 0.f;
  for (int h = 0; h < H; ++h)
    for (int w = 0; w < W; ++w)
      s += (float)in[(((size_t)b * H + h) * W + w) * C + c];
  out[idx] = s / (float)(H * W);
}

// ---------------------------------------------------------------------------
// Small dense layers
// ---------------------------------------------------------------------------
__global__ void k_linear(const float* __restrict__ in, const float* __restrict__ w,
                         const float* __restrict__ bias, float* __restrict__ out,
                         int B, int I, int O, int relu) {
  int idx = blockIdx.x * blockDim.x + threadIdx.x;
  if (idx >= B * O) return;
  int o = idx % O;
  int b = idx / O;
  float acc = bias[o];
  for (int i = 0; i < I; ++i) acc += in[(size_t)b * I + i] * w[(size_t)o * I + i];
  if (relu) acc = fmaxf(acc, 0.f);
  out[idx] = acc;
}

// ---------------------------------------------------------------------------
// 2-qubit circuit per (b,g)
// ---------------------------------------------------------------------------
__global__ void k_quantum(const float* __restrict__ h, const float* __restrict__ qw,
                          float* __restrict__ q, int B) {
  int idx = blockIdx.x * blockDim.x + threadIdx.x;
  if (idx >= B * 64) return;
  int g = idx & 63;
  int b = idx >> 6;
  float s00 = 1.f, s01 = 0.f, s10 = 0.f, s11 = 0.f;
  float t, c, s;

  t = 0.5f * h[b * 128 + g * 2 + 0]; c = cosf(t); s = sinf(t);
  { float a0 = c * s00 - s * s10, a1 = c * s01 - s * s11;
    float b0 = s * s00 + c * s10, b1 = s * s01 + c * s11;
    s00 = a0; s01 = a1; s10 = b0; s11 = b1; }
  t = 0.5f * h[b * 128 + g * 2 + 1]; c = cosf(t); s = sinf(t);
  { float a0 = c * s00 - s * s01, a1 = c * s10 - s * s11;
    float b0 = s * s00 + c * s01, b1 = s * s10 + c * s11;
    s00 = a0; s10 = a1; s01 = b0; s11 = b1; }

  for (int l = 0; l < 6; ++l) {
    t = 0.5f * qw[(g * 6 + l) * 2 + 0]; c = cosf(t); s = sinf(t);
    { float a0 = c * s00 - s * s10, a1 = c * s01 - s * s11;
      float b0 = s * s00 + c * s10, b1 = s * s01 + c * s11;
      s00 = a0; s01 = a1; s10 = b0; s11 = b1; }
    t = 0.5f * qw[(g * 6 + l) * 2 + 1]; c = cosf(t); s = sinf(t);
    { float a0 = c * s00 - s * s01, a1 = c * s10 - s * s11;
      float b0 = s * s00 + c * s01, b1 = s * s10 + c * s11;
      s00 = a0; s10 = a1; s01 = b0; s11 = b1; }
    float tmp = s10; s10 = s11; s11 = tmp;      // CNOT(0 -> 1)
  }
  float p00 = s00 * s00, p01 = s01 * s01, p10 = s10 * s10, p11 = s11 * s11;
  q[b * 128 + g * 2 + 0] = p00 + p01 - p10 - p11;
  q[b * 128 + g * 2 + 1] = p00 - p01 + p10 - p11;
}

// ---------------------------------------------------------------------------
static inline unsigned cdiv_sz(size_t a, size_t b) { return (unsigned)((a + b - 1) / b); }

extern "C" void kernel_launch(void* const* d_in, const int* in_sizes, int n_in,
                              void* d_out, int out_size, void* d_ws, size_t ws_size,
                              hipStream_t stream) {
  (void)in_sizes; (void)n_in; (void)out_size; (void)ws_size;
  const float* x = (const float*)d_in[0];

  // Input leaves (depth-first insertion order): 0:x, 1:conv1, 2-5:bn1, layers, head.
  struct CSpec { int w, bn, O, I, K; };
  static const CSpec cs[20] = {
      {1, 2, 64, 3, 7},
      {6, 7, 64, 64, 3},   {11, 12, 64, 64, 3},  {16, 17, 64, 64, 3},  {21, 22, 64, 64, 3},
      {26, 27, 128, 64, 3},{31, 32, 128, 128, 3},{36, 37, 128, 64, 1}, {41, 42, 128, 128, 3},
      {46, 47, 128, 128, 3},
      {51, 52, 256, 128, 3},{56, 57, 256, 256, 3},{61, 62, 256, 128, 1},{66, 67, 256, 256, 3},
      {71, 72, 256, 256, 3},
      {76, 77, 512, 256, 3},{81, 82, 512, 512, 3},{86, 87, 512, 256, 1},{91, 92, 512, 512, 3},
      {96, 97, 512, 512, 3},
  };

  char* ws = (char*)d_ws;
  size_t cur = 0;
  auto walloc = [&](size_t bytes) -> void* {
    void* p = ws + cur;
    cur = (cur + bytes + 255) & ~(size_t)255;
    return p;
  };

  _Float16* wt[20]; float* sc[20]; float* sh[20];
  for (int i = 0; i < 20; ++i) {
    const CSpec& c = cs[i];
    size_t we = (size_t)c.O * c.I * c.K * c.K;
    wt[i] = (_Float16*)walloc(we * 2);
    sc[i] = (float*)walloc((size_t)c.O * 4);
    sh[i] = (float*)walloc((size_t)c.O * 4);
    k_wconv<<<cdiv_sz(we, 256), 256, 0, stream>>>((const float*)d_in[c.w], wt[i], c.O, c.I, c.K, c.K);
    k_bnfold<<<cdiv_sz(c.O, 256), 256, 0, stream>>>(
        (const float*)d_in[c.bn], (const float*)d_in[c.bn + 1],
        (const float*)d_in[c.bn + 2], (const float*)d_in[c.bn + 3], sc[i], sh[i], c.O);
  }

  _Float16* xh   = (_Float16*)walloc((size_t)64 * 224 * 224 * 3 * 2);
  _Float16* bufA = (_Float16*)walloc((size_t)64 * 112 * 112 * 64 * 2);
  size_t pbytes  = (size_t)64 * 56 * 56 * 64 * 2;
  _Float16* p0 = (_Float16*)walloc(pbytes);
  _Float16* p1 = (_Float16*)walloc(pbytes);
  _Float16* p2 = (_Float16*)walloc(pbytes);
  _Float16* p3 = (_Float16*)walloc(pbytes);
  float* feat = (float*)walloc((size_t)64 * 512 * 4);
  float* hb   = (float*)walloc((size_t)64 * 128 * 4);
  float* qb   = (float*)walloc((size_t)64 * 128 * 4);
  float* h2   = (float*)walloc((size_t)64 * 128 * 4);
  float* h3   = (float*)walloc((size_t)64 * 128 * 4);

  auto conv = [&](int i, const _Float16* in, const _Float16* res, _Float16* out,
                  int N, int Hin, int Win, int Cin, int stride, int pad, int relu) {
    const CSpec& c = cs[i];
    int Hout = (Hin + 2 * pad - c.K) / stride + 1;
    int Wout = (Win + 2 * pad - c.K) / stride + 1;
    dim3 grid(cdiv_sz((size_t)N * Hout * Wout, 64), (unsigned)(c.O / 64));
    k_conv_wmma<<<grid, 128, 0, stream>>>(in, wt[i], sc[i], sh[i], res, out,
                                          N, Hin, Win, Cin, c.O, Hout, Wout,
                                          c.K, c.K, stride, pad, relu);
  };

  // ---- stem ----
  k_nchw2nhwc<<<cdiv_sz((size_t)64 * 3 * 224 * 224, 256), 256, 0, stream>>>(x, xh, 64, 3, 224, 224);
  k_conv_stem<<<cdiv_sz((size_t)64 * 112 * 112, 16), 128, 0, stream>>>(xh, wt[0], sc[0], sh[0], bufA, 64);
  k_maxpool<<<cdiv_sz((size_t)64 * 56 * 56 * 64, 256), 256, 0, stream>>>(bufA, p0, 64, 64, 112, 112, 56, 56);

  // ---- layer1 ----
  conv(1, p0, nullptr, p1, 64, 56, 56, 64, 1, 1, 1);
  conv(2, p1, p0,      p2, 64, 56, 56, 64, 1, 1, 1);
  conv(3, p2, nullptr, p0, 64, 56, 56, 64, 1, 1, 1);
  conv(4, p0, p2,      p1, 64, 56, 56, 64, 1, 1, 1);
  // ---- layer2 ----
  conv(5, p1, nullptr, p0, 64, 56, 56, 64, 2, 1, 1);                  // -> 28x28x128
  conv(7, p1, nullptr, p2, 64, 56, 56, 64, 2, 0, 0);                  // downsample
  conv(6, p0, p2,      p3, 64, 28, 28, 128, 1, 1, 1);
  conv(8, p3, nullptr, p0, 64, 28, 28, 128, 1, 1, 1);
  conv(9, p0, p3,      p1, 64, 28, 28, 128, 1, 1, 1);
  // ---- layer3 ----
  conv(10, p1, nullptr, p0, 64, 28, 28, 128, 2, 1, 1);                // -> 14x14x256
  conv(12, p1, nullptr, p2, 64, 28, 28, 128, 2, 0, 0);
  conv(11, p0, p2,      p3, 64, 14, 14, 256, 1, 1, 1);
  conv(13, p3, nullptr, p0, 64, 14, 14, 256, 1, 1, 1);
  conv(14, p0, p3,      p1, 64, 14, 14, 256, 1, 1, 1);
  // ---- layer4 ----
  conv(15, p1, nullptr, p0, 64, 14, 14, 256, 2, 1, 1);                // -> 7x7x512
  conv(17, p1, nullptr, p2, 64, 14, 14, 256, 2, 0, 0);
  conv(16, p0, p2,      p3, 64, 7, 7, 512, 1, 1, 1);
  conv(18, p3, nullptr, p0, 64, 7, 7, 512, 1, 1, 1);
  conv(19, p0, p3,      p1, 64, 7, 7, 512, 1, 1, 1);

  // ---- head ----
  k_avgpool<<<cdiv_sz(64 * 512, 256), 256, 0, stream>>>(p1, feat, 64, 512, 7, 7);
  k_linear<<<cdiv_sz(64 * 128, 128), 128, 0, stream>>>(feat, (const float*)d_in[101],
                                                       (const float*)d_in[102], hb, 64, 512, 128, 0);
  k_quantum<<<cdiv_sz(64 * 64, 128), 128, 0, stream>>>(hb, (const float*)d_in[103], qb, 64);
  k_linear<<<cdiv_sz(64 * 128, 128), 128, 0, stream>>>(qb, (const float*)d_in[104],
                                                       (const float*)d_in[105], h2, 64, 128, 128, 1);
  k_linear<<<cdiv_sz(64 * 128, 128), 128, 0, stream>>>(h2, (const float*)d_in[106],
                                                       (const float*)d_in[107], h3, 64, 128, 128, 1);
  k_linear<<<cdiv_sz(64 * 10, 64), 64, 0, stream>>>(h3, (const float*)d_in[108],
                                                    (const float*)d_in[109], (float*)d_out, 64, 128, 10, 0);
}
